// OnlineMemoryAttention_13537736917790
// MI455X (gfx1250) — matrix operand order repaired and checked by reading, not compiled, not run on hardware
//
#include <hip/hip_runtime.h>
#include <hip/hip_bf16.h>

typedef __attribute__((ext_vector_type(16))) __bf16 v16bf;
typedef __attribute__((ext_vector_type(8)))  float  v8f;

#define B_   2
#define T_   2048
#define D_   512
#define H_   8
#define DH_  64
#define SCALE_ 0.125f      /* 64^-0.5 */
#define DECAY_ 0.99f
#define NEG_  -1e30f
#define CH_  128           /* scan chunk */
#define NC_  (T_ / CH_)    /* 16 chunks  */

// ---------------------------------------------------------------- helpers ----
static __device__ __forceinline__ v8f zero8() {
    v8f r;
#pragma unroll
    for (int i = 0; i < 8; i++) r[i] = 0.f;
    return r;
}
static __device__ __forceinline__ v8f splat8(float x) {
    v8f r;
#pragma unroll
    for (int i = 0; i < 8; i++) r[i] = x;
    return r;
}

// A-matrix 16x32 bf16 fragment: two 8-element K runs per lane.
// lanes 0-15: K = {kk..kk+7, kk+16..kk+23}; lanes 16-31: +8.
static __device__ __forceinline__ v16bf a_frag_bf(const __bf16* row, int kk, int hh) {
    const __bf16* p0 = row + kk + (hh ? 8 : 0);
    const __bf16* p1 = p0 + 16;
    v16bf r;
#pragma unroll
    for (int i = 0; i < 8; i++) { r[i] = p0[i]; r[i + 8] = p1[i]; }
    return r;
}
// B-matrix 32x16 bf16 fragment: two 8-element K runs per lane.
// lanes 0-15: K = {kk..kk+7, kk+16..kk+23}; lanes 16-31: +8.
static __device__ __forceinline__ v16bf b_frag_bf(const __bf16* row, int kk, int hh) {
    const __bf16* p = row + kk + (hh ? 8 : 0);
    v16bf r;
#pragma unroll
    for (int i = 0; i < 8; i++) { r[i] = p[i]; r[i + 8] = p[i + 16]; }
    return r;
}
static __device__ __forceinline__ v8f wmma_bf(v16bf a, v16bf b, v8f c) {
    return __builtin_amdgcn_wmma_f32_16x16x32_bf16(false, a, false, b, (short)0, c,
                                                   false, false);
}
static __device__ __forceinline__ float sigmoidf_(float x) {
    return 1.f / (1.f + __expf(-x));
}
// lane -> (k,l) pair among 21 unique symmetric 6x6 pairs
static __device__ __forceinline__ void pair21(int lane, int& k, int& l) {
    k = 0; l = 0;
    int cnt = 0;
#pragma unroll
    for (int a = 0; a < 6; a++)
#pragma unroll
        for (int c = a; c < 6; c++) {
            if (cnt == lane) { k = a; l = c; }
            cnt++;
        }
}

// ------------------------------------------------ f32 -> bf16 pre-pass -------
__global__ void cvt_bf16_kernel(const float* __restrict__ src, __bf16* __restrict__ dst,
                                int n) {
    int i = blockIdx.x * blockDim.x + threadIdx.x;
    if (i < n) dst[i] = (__bf16)src[i];
}

// ------------------------------------------------ generic bf16 WMMA GEMM -----
// C[M,N] = A[M,K] @ W[N,K]^T (+bias). One wave per 16x64 C block (4 N-tiles
// share one A fragment), 4 waves / block. Pure load+wmma inner loop.
__global__ void gemm_bf_kernel(const __bf16* __restrict__ A, const __bf16* __restrict__ W,
                               const float* __restrict__ bias, float* __restrict__ C,
                               int M, int N, int K, int ldc) {
    int wave = threadIdx.x >> 5;
    int lane = threadIdx.x & 31;
    int hh = lane >> 4, idx = lane & 15;
    int ng = (N + 63) >> 6;
    int tile = blockIdx.x * (blockDim.x >> 5) + wave;
    int tm = tile / ng, tg = tile - tm * ng;
    if (tm * 16 >= M) return;
    const __bf16* Arow = A + (size_t)(tm * 16 + idx) * K;
    int c0 = tg * 64 + 0 * 16 + idx; if (c0 >= N) c0 = N - 1;
    int c1 = tg * 64 + 1 * 16 + idx; if (c1 >= N) c1 = N - 1;
    int c2 = tg * 64 + 2 * 16 + idx; if (c2 >= N) c2 = N - 1;
    int c3 = tg * 64 + 3 * 16 + idx; if (c3 >= N) c3 = N - 1;
    const __bf16* W0 = W + (size_t)c0 * K;
    const __bf16* W1 = W + (size_t)c1 * K;
    const __bf16* W2 = W + (size_t)c2 * K;
    const __bf16* W3 = W + (size_t)c3 * K;
    v8f acc[4] = {zero8(), zero8(), zero8(), zero8()};
    for (int kk = 0; kk < K; kk += 32) {
        __builtin_prefetch(Arow + kk + 64, 0, 0);
        __builtin_prefetch(W0 + kk + 64, 0, 0);
        v16bf a = a_frag_bf(Arow, kk, hh);
        acc[0] = wmma_bf(a, b_frag_bf(W0, kk, hh), acc[0]);
        acc[1] = wmma_bf(a, b_frag_bf(W1, kk, hh), acc[1]);
        acc[2] = wmma_bf(a, b_frag_bf(W2, kk, hh), acc[2]);
        acc[3] = wmma_bf(a, b_frag_bf(W3, kk, hh), acc[3]);
    }
#pragma unroll
    for (int j = 0; j < 4; j++) {
        int n = tg * 64 + j * 16 + idx;
        if (n < N) {
            float bs = bias ? bias[n] : 0.f;
#pragma unroll
            for (int r = 0; r < 8; r++) {
                int m = tm * 16 + r + hh * 8;
                if (m < M) C[(size_t)m * ldc + n] = acc[j][r] + bs;
            }
        }
    }
}

// ------------------------------------------------ QKV GEMM + scatter ---------
// qkv = xb @ qkv_wb^T + qkv_b -> q[B,H,T,DH], k[B,H,T,DH], vT[B,H,DH,T] bf16
__global__ void gemm_qkv_kernel(const __bf16* __restrict__ X, const __bf16* __restrict__ W,
                                const float* __restrict__ bias,
                                __bf16* __restrict__ qb, __bf16* __restrict__ kb,
                                __bf16* __restrict__ vtb) {
    const int M = B_ * T_, N = 3 * D_, K = D_;
    int wave = threadIdx.x >> 5;
    int lane = threadIdx.x & 31;
    int hh = lane >> 4, idx = lane & 15;
    int ng = N >> 6;
    int tile = blockIdx.x * (blockDim.x >> 5) + wave;
    int tm = tile / ng, tg = tile - tm * ng;
    if (tm * 16 >= M) return;
    const __bf16* Arow = X + (size_t)(tm * 16 + idx) * K;
    const __bf16* W0 = W + (size_t)(tg * 64 + 0 * 16 + idx) * K;
    const __bf16* W1 = W + (size_t)(tg * 64 + 1 * 16 + idx) * K;
    const __bf16* W2 = W + (size_t)(tg * 64 + 2 * 16 + idx) * K;
    const __bf16* W3 = W + (size_t)(tg * 64 + 3 * 16 + idx) * K;
    v8f acc[4] = {zero8(), zero8(), zero8(), zero8()};
    for (int kk = 0; kk < K; kk += 32) {
        __builtin_prefetch(Arow + kk + 64, 0, 0);
        __builtin_prefetch(W0 + kk + 64, 0, 0);
        v16bf a = a_frag_bf(Arow, kk, hh);
        acc[0] = wmma_bf(a, b_frag_bf(W0, kk, hh), acc[0]);
        acc[1] = wmma_bf(a, b_frag_bf(W1, kk, hh), acc[1]);
        acc[2] = wmma_bf(a, b_frag_bf(W2, kk, hh), acc[2]);
        acc[3] = wmma_bf(a, b_frag_bf(W3, kk, hh), acc[3]);
    }
#pragma unroll
    for (int j = 0; j < 4; j++) {
        int n = tg * 64 + j * 16 + idx;
        float bs = bias[n];
        int sect = n >> 9;           // 0=q 1=k 2=v
        int within = n & 511;
        int h = within >> 6, dh = within & 63;
#pragma unroll
        for (int r = 0; r < 8; r++) {
            int m = tm * 16 + r + hh * 8;
            int b = m >> 11, t = m & (T_ - 1);
            float val = acc[j][r] + bs;
            size_t bh = (size_t)(b * H_ + h);
            if (sect == 0)      qb[(bh * T_ + t) * DH_ + dh] = (__bf16)val;
            else if (sect == 1) kb[(bh * T_ + t) * DH_ + dh] = (__bf16)val;
            else                vtb[(bh * DH_ + dh) * T_ + t] = (__bf16)val;
        }
    }
}

// ------------------------------------------------ online softmax step --------
static __device__ __forceinline__ void softmax_step(v8f& s0, v8f& s1, v8f& mrow,
                                                    v8f& lrow, v8f& alpha,
                                                    int qbase, int k0, int idx, int hh) {
    int key0 = k0 + idx, key1 = k0 + 16 + idx;
    v8f mnew = zero8();
#pragma unroll
    for (int r = 0; r < 8; r++) {
        int qr = qbase + r + hh * 8;
        float a = (key0 <= qr) ? s0[r] * SCALE_ : NEG_;
        float c = (key1 <= qr) ? s1[r] * SCALE_ : NEG_;
        s0[r] = a; s1[r] = c;
        float v = fmaxf(a, c);
        v = fmaxf(v, __shfl_xor(v, 1));
        v = fmaxf(v, __shfl_xor(v, 2));
        v = fmaxf(v, __shfl_xor(v, 4));
        v = fmaxf(v, __shfl_xor(v, 8));
        mnew[r] = fmaxf(mrow[r], v);
    }
#pragma unroll
    for (int r = 0; r < 8; r++) {
        alpha[r] = __expf(mrow[r] - mnew[r]);
        float p0 = __expf(s0[r] - mnew[r]);
        float p1 = __expf(s1[r] - mnew[r]);
        s0[r] = p0; s1[r] = p1;
        float v = p0 + p1;
        v += __shfl_xor(v, 1);
        v += __shfl_xor(v, 2);
        v += __shfl_xor(v, 4);
        v += __shfl_xor(v, 8);
        lrow[r] = lrow[r] * alpha[r] + v;
        mrow[r] = mnew[r];
    }
}

// ------------------------------------------------ flash attention (WMMA) -----
// grid: (T/32, B*H), block: 32. One wave = two 16-query tiles sharing K/V frags.
__global__ void attn_kernel(const __bf16* __restrict__ qb, const __bf16* __restrict__ kb,
                            const __bf16* __restrict__ vtb, float* __restrict__ seq) {
    __shared__ __bf16 PshA[16][40];
    __shared__ __bf16 PshB[16][40];
    int lane = threadIdx.x & 31;
    int hh = lane >> 4, idx = lane & 15;
    int bh = blockIdx.y;
    int b = bh >> 3, h = bh & 7;
    int q0 = blockIdx.x * 32;
    const __bf16* Q  = qb  + (size_t)bh * T_ * DH_;
    const __bf16* Kp = kb  + (size_t)bh * T_ * DH_;
    const __bf16* VT = vtb + (size_t)bh * DH_ * T_;

    const __bf16* QrowA = Q + (size_t)(q0 + idx) * DH_;
    const __bf16* QrowB = Q + (size_t)(q0 + 16 + idx) * DH_;
    v16bf qA0 = a_frag_bf(QrowA, 0, hh), qA1 = a_frag_bf(QrowA, 32, hh);
    v16bf qB0 = a_frag_bf(QrowB, 0, hh), qB1 = a_frag_bf(QrowB, 32, hh);

    v8f oA[4] = {zero8(), zero8(), zero8(), zero8()};
    v8f oB[4] = {zero8(), zero8(), zero8(), zero8()};
    v8f mA = splat8(NEG_), lA = zero8();
    v8f mB = splat8(NEG_), lB = zero8();

    for (int k0 = 0; k0 < q0 + 32; k0 += 32) {
        const __bf16* Kr0 = Kp + (size_t)(k0 + idx) * DH_;
        const __bf16* Kr1 = Kp + (size_t)(k0 + 16 + idx) * DH_;
        __builtin_prefetch(Kr0 + 32 * DH_, 0, 0);
        __builtin_prefetch(Kr1 + 32 * DH_, 0, 0);
        v16bf kf00 = b_frag_bf(Kr0, 0, hh), kf01 = b_frag_bf(Kr0, 32, hh);
        v16bf kf10 = b_frag_bf(Kr1, 0, hh), kf11 = b_frag_bf(Kr1, 32, hh);

        v8f sA0 = zero8(), sA1 = zero8(), sB0 = zero8(), sB1 = zero8();
        sA0 = wmma_bf(qA0, kf00, sA0); sA0 = wmma_bf(qA1, kf01, sA0);
        sA1 = wmma_bf(qA0, kf10, sA1); sA1 = wmma_bf(qA1, kf11, sA1);
        sB0 = wmma_bf(qB0, kf00, sB0); sB0 = wmma_bf(qB1, kf01, sB0);
        sB1 = wmma_bf(qB0, kf10, sB1); sB1 = wmma_bf(qB1, kf11, sB1);

        v8f aA = zero8(), aB = zero8();
        softmax_step(sA0, sA1, mA, lA, aA, q0, k0, idx, hh);
        softmax_step(sB0, sB1, mB, lB, aB, q0 + 16, k0, idx, hh);
#pragma unroll
        for (int j = 0; j < 4; j++) { oA[j] = oA[j] * aA; oB[j] = oB[j] * aB; }

        // re-layout probs C-layout -> A-fragment through LDS
#pragma unroll
        for (int r = 0; r < 8; r++) {
            PshA[r + hh * 8][idx]      = (__bf16)sA0[r];
            PshA[r + hh * 8][idx + 16] = (__bf16)sA1[r];
            PshB[r + hh * 8][idx]      = (__bf16)sB0[r];
            PshB[r + hh * 8][idx + 16] = (__bf16)sB1[r];
        }
        __syncthreads();
        v16bf pfA = a_frag_bf(&PshA[idx][0], 0, hh);
        v16bf pfB = a_frag_bf(&PshB[idx][0], 0, hh);
#pragma unroll
        for (int j = 0; j < 4; j++) {
            v16bf vf = b_frag_bf(VT + (size_t)(j * 16 + idx) * T_, k0, hh);
            oA[j] = wmma_bf(pfA, vf, oA[j]);
            oB[j] = wmma_bf(pfB, vf, oB[j]);
        }
    }
#pragma unroll
    for (int r = 0; r < 8; r++) {
        float invA = 1.f / lA[r];
        float invB = 1.f / lB[r];
        int qrA = q0 + r + hh * 8;
        int qrB = qrA + 16;
        float* outA = seq + ((size_t)(b * T_ + qrA)) * D_ + h * DH_;
        float* outB = seq + ((size_t)(b * T_ + qrB)) * D_ + h * DH_;
#pragma unroll
        for (int j = 0; j < 4; j++) {
            outA[j * 16 + idx] = oA[j][r] * invA;
            outB[j * 16 + idx] = oB[j][r] * invB;
        }
    }
}

// ------------------------------------------------ Plucker lines + J ----------
__global__ void lines_kernel(const float* __restrict__ w1p, const float* __restrict__ w2p,
                             const float* __restrict__ r1p, const float* __restrict__ r2p,
                             float* __restrict__ uarr, float* __restrict__ rarr) {
    int tid = blockIdx.x * blockDim.x + threadIdx.x;
    if (tid >= B_ * T_ * H_) return;
    int h = tid & 7;
    int t = (tid >> 3) & (T_ - 1);
    int b = tid >> 14;
    int m = b * T_ + t;

    float a0 = 0, a1 = 0, a2 = 0, a3 = 0;
    if (t > 0) {
        const float* p = w1p + (size_t)(m - 1) * 32 + h * 4;
        a0 = p[0]; a1 = p[1]; a2 = p[2]; a3 = p[3];
    }
    const float* q = w2p + (size_t)m * 32 + h * 4;
    float b0 = q[0], b1 = q[1], b2 = q[2], b3 = q[3];
    float L0 = a0 * b1 - a1 * b0, L1 = a0 * b2 - a2 * b0, L2 = a0 * b3 - a3 * b0;
    float L3 = a1 * b2 - a2 * b1, L4 = a1 * b3 - a3 * b1, L5 = a2 * b3 - a3 * b2;
    float nrm = sqrtf(L0 * L0 + L1 * L1 + L2 * L2 + L3 * L3 + L4 * L4 + L5 * L5);
    float inv = 1.f / fmaxf(nrm, 1e-12f);
    float* u = uarr + ((size_t)(b * H_ + h) * T_ + t) * 6;
    u[0] =  L5 * inv; u[1] = -L4 * inv; u[2] =  L3 * inv;   // J * w
    u[3] =  L2 * inv; u[4] = -L1 * inv; u[5] =  L0 * inv;

    const float* pr = r1p + (size_t)m * 32 + h * 4;
    const float* qr = r2p + (size_t)m * 32 + h * 4;
    a0 = pr[0]; a1 = pr[1]; a2 = pr[2]; a3 = pr[3];
    b0 = qr[0]; b1 = qr[1]; b2 = qr[2]; b3 = qr[3];
    L0 = a0 * b1 - a1 * b0; L1 = a0 * b2 - a2 * b0; L2 = a0 * b3 - a3 * b0;
    L3 = a1 * b2 - a2 * b1; L4 = a1 * b3 - a3 * b1; L5 = a2 * b3 - a3 * b2;
    nrm = sqrtf(L0 * L0 + L1 * L1 + L2 * L2 + L3 * L3 + L4 * L4 + L5 * L5);
    inv = 1.f / fmaxf(nrm, 1e-12f);
    float* rr = rarr + ((size_t)(b * H_ + h) * T_ + t) * 6;
    rr[0] = L0 * inv; rr[1] = L1 * inv; rr[2] = L2 * inv;
    rr[3] = L3 * inv; rr[4] = L4 * inv; rr[5] = L5 * inv;
}

// ------------------------------------------------ blocked decay scan ---------
// S_{t+1} = DECAY*(S_t + u_t (x) u_t); score_t = sum_{k<=l} c_kl r_tk r_tl S_t,kl
__global__ void scan_pass1(const float* __restrict__ uarr, float* __restrict__ Pst) {
    int bh = blockIdx.y, c = blockIdx.x, lane = threadIdx.x;
    bool act = lane < 21;
    int k, l; pair21(lane, k, l);
    const float* ub = uarr + ((size_t)bh * T_ + c * CH_) * 6;
    float R = 0.f;
#pragma unroll 4
    for (int t = 0; t < CH_; t++) {
        float uk = 0, ul = 0;
        if (act) { uk = ub[t * 6 + k]; ul = ub[t * 6 + l]; }
        R = DECAY_ * (R + uk * ul);
    }
    Pst[((size_t)bh * NC_ + c) * 32 + lane] = R;
}
__global__ void scan_pass2(const float* __restrict__ Pst, float* __restrict__ Ast) {
    int bh = blockIdx.x, lane = threadIdx.x;
    float dC = __powf(DECAY_, (float)CH_);
    float A = 0.f;
    for (int c = 0; c < NC_; c++) {
        Ast[((size_t)bh * NC_ + c) * 32 + lane] = A;
        A = dC * A + Pst[((size_t)bh * NC_ + c) * 32 + lane];
    }
}
__global__ void scan_pass3(const float* __restrict__ uarr, const float* __restrict__ rarr,
                           const float* __restrict__ Ast, float* __restrict__ score) {
    int bh = blockIdx.y, c = blockIdx.x, lane = threadIdx.x;
    bool act = lane < 21;
    int k, l; pair21(lane, k, l);
    float coef = (k == l) ? 1.f : 2.f;
    const float* ub = uarr + ((size_t)bh * T_ + c * CH_) * 6;
    const float* rb = rarr + ((size_t)bh * T_ + c * CH_) * 6;
    float* sb = score + (size_t)bh * T_ + c * CH_;
    float S = Ast[((size_t)bh * NC_ + c) * 32 + lane];
#pragma unroll 4
    for (int t = 0; t < CH_; t++) {
        float uk = 0, ul = 0, rk = 0, rl = 0;
        if (act) {
            uk = ub[t * 6 + k]; ul = ub[t * 6 + l];
            rk = rb[t * 6 + k]; rl = rb[t * 6 + l];
        }
        float part = act ? coef * rk * rl * S : 0.f;
        part += __shfl_xor(part, 1);
        part += __shfl_xor(part, 2);
        part += __shfl_xor(part, 4);
        part += __shfl_xor(part, 8);
        part += __shfl_xor(part, 16);
        if (lane == 0) sb[t] = part;
        S = DECAY_ * (S + uk * ul);
    }
}

// ------------------------------------------------ gating ---------------------
__global__ void gate_kernel(const float* __restrict__ score, const float* __restrict__ glin,
                            const float* __restrict__ mem_scale, float* __restrict__ gated) {
    int m = blockIdx.x * blockDim.x + threadIdx.x;
    if (m >= B_ * T_) return;
    int b = m / T_, t = m % T_;
    float acc = 0.f;
#pragma unroll
    for (int h = 0; h < H_; h++) {
        float s = score[((size_t)(b * H_ + h)) * T_ + t];
        float g = glin[(size_t)m * H_ + h];
        acc += sigmoidf_(s * mem_scale[h]) * sigmoidf_(g);
    }
    gated[m] = acc * (1.f / H_);
}

// ------------------------------------------------ fuse -> bf16 ---------------
__global__ void combine_kernel(const float* __restrict__ seq, const float* __restrict__ memval,
                               const float* __restrict__ gated, __bf16* __restrict__ comb) {
    size_t i = (size_t)blockIdx.x * blockDim.x + threadIdx.x;
    if (i >= (size_t)B_ * T_ * D_) return;
    size_t m = i / D_;
    comb[i] = (__bf16)(seq[i] + gated[m] * memval[i]);
}

// =============================================================== launcher ====
extern "C" void kernel_launch(void* const* d_in, const int* in_sizes, int n_in,
                              void* d_out, int out_size, void* d_ws, size_t ws_size,
                              hipStream_t stream) {
    const float* x           = (const float*)d_in[0];
    const float* qkv_w       = (const float*)d_in[1];
    const float* qkv_b       = (const float*)d_in[2];
    const float* w1_write    = (const float*)d_in[3];
    const float* w2_write    = (const float*)d_in[4];
    const float* w1_read     = (const float*)d_in[5];
    const float* w2_read     = (const float*)d_in[6];
    const float* mem_value_w = (const float*)d_in[7];
    const float* mem_value_b = (const float*)d_in[8];
    const float* mem_gate_w  = (const float*)d_in[9];
    const float* mem_gate_b  = (const float*)d_in[10];
    const float* mem_scale   = (const float*)d_in[11];
    const float* out_w       = (const float*)d_in[12];
    const float* out_b       = (const float*)d_in[13];
    float* out = (float*)d_out;

    const int M = B_ * T_;                      // 4096
    char* ws = (char*)d_ws;
    size_t off = 0;
    auto alloc = [&](size_t bytes) -> void* {
        void* p = ws + off;
        off = (off + bytes + 255) & ~(size_t)255;
        return p;
    };
    __bf16* qb     = (__bf16*)alloc((size_t)B_ * H_ * T_ * DH_ * 2);
    __bf16* kb     = (__bf16*)alloc((size_t)B_ * H_ * T_ * DH_ * 2);
    __bf16* vtb    = (__bf16*)alloc((size_t)B_ * H_ * DH_ * T_ * 2);
    float* w1p     = (float*)alloc((size_t)M * 32 * 4);
    float* w2p     = (float*)alloc((size_t)M * 32 * 4);
    float* r1p     = (float*)alloc((size_t)M * 32 * 4);
    float* r2p     = (float*)alloc((size_t)M * 32 * 4);
    float* uarr    = (float*)alloc((size_t)B_ * H_ * T_ * 6 * 4);
    float* rarr    = (float*)alloc((size_t)B_ * H_ * T_ * 6 * 4);
    float* memval  = (float*)alloc((size_t)M * D_ * 4);
    float* glin    = (float*)alloc((size_t)M * H_ * 4);
    float* mscore  = (float*)alloc((size_t)B_ * H_ * T_ * 4);
    float* gated   = (float*)alloc((size_t)M * 4);
    float* seq     = (float*)alloc((size_t)M * D_ * 4);
    __bf16* comb   = (__bf16*)alloc((size_t)M * D_ * 2);
    float* Pst     = (float*)alloc((size_t)B_ * H_ * NC_ * 32 * 4);
    float* Ast     = (float*)alloc((size_t)B_ * H_ * NC_ * 32 * 4);
    // bf16 copies of x and all weights (converted once per launch)
    __bf16* xb     = (__bf16*)alloc((size_t)M * D_ * 2);
    __bf16* qkvwb  = (__bf16*)alloc((size_t)3 * D_ * D_ * 2);
    __bf16* mvwb   = (__bf16*)alloc((size_t)D_ * D_ * 2);
    __bf16* owb    = (__bf16*)alloc((size_t)D_ * D_ * 2);
    __bf16* w1wb   = (__bf16*)alloc((size_t)4 * H_ * D_ * 2);
    __bf16* w2wb   = (__bf16*)alloc((size_t)4 * H_ * D_ * 2);
    __bf16* r1wb   = (__bf16*)alloc((size_t)4 * H_ * D_ * 2);
    __bf16* r2wb   = (__bf16*)alloc((size_t)4 * H_ * D_ * 2);
    __bf16* gwb    = (__bf16*)alloc((size_t)H_ * D_ * 2);

    auto cvt = [&](const float* s, __bf16* d, int n) {
        cvt_bf16_kernel<<<(n + 255) / 256, 256, 0, stream>>>(s, d, n);
    };
    auto gemm = [&](const __bf16* A, const __bf16* W, const float* bias, float* C,
                    int m, int n, int k, int ldc) {
        int tiles = ((m + 15) / 16) * ((n + 63) / 64);
        int blocks = (tiles + 3) / 4;
        gemm_bf_kernel<<<blocks, 128, 0, stream>>>(A, W, bias, C, m, n, k, ldc);
    };

    // 0) one-time bf16 conversion of activations + weights
    cvt(x, xb, M * D_);
    cvt(qkv_w, qkvwb, 3 * D_ * D_);
    cvt(mem_value_w, mvwb, D_ * D_);
    cvt(out_w, owb, D_ * D_);
    cvt(w1_write, w1wb, 4 * H_ * D_);
    cvt(w2_write, w2wb, 4 * H_ * D_);
    cvt(w1_read,  r1wb, 4 * H_ * D_);
    cvt(w2_read,  r2wb, 4 * H_ * D_);
    cvt(mem_gate_w, gwb, H_ * D_);

    // 1) QKV projection (WMMA) -> q/k/vT bf16
    {
        int tiles = (M / 16) * ((3 * D_) / 64);
        gemm_qkv_kernel<<<(tiles + 3) / 4, 128, 0, stream>>>(xb, qkvwb, qkv_b,
                                                             qb, kb, vtb);
    }
    // 2) mem_val / line projections / gate linear (WMMA)
    gemm(xb, mvwb, mem_value_b, memval, M, D_, D_, D_);
    gemm(xb, w1wb, nullptr, w1p, M, 32, D_, 32);
    gemm(xb, w2wb, nullptr, w2p, M, 32, D_, 32);
    gemm(xb, r1wb, nullptr, r1p, M, 32, D_, 32);
    gemm(xb, r2wb, nullptr, r2p, M, 32, D_, 32);
    gemm(xb, gwb, mem_gate_b, glin, M, H_, D_, H_);
    // 3) Plucker lines + J
    lines_kernel<<<(B_ * T_ * H_ + 255) / 256, 256, 0, stream>>>(w1p, w2p, r1p, r2p,
                                                                 uarr, rarr);
    // 4) causal flash attention (WMMA), 32 queries per wave
    attn_kernel<<<dim3(T_ / 32, B_ * H_), 32, 0, stream>>>(qb, kb, vtb, seq);
    // 5) blocked decay scan -> mem_score
    scan_pass1<<<dim3(NC_, B_ * H_), 32, 0, stream>>>(uarr, Pst);
    scan_pass2<<<B_ * H_, 32, 0, stream>>>(Pst, Ast);
    scan_pass3<<<dim3(NC_, B_ * H_), 32, 0, stream>>>(uarr, rarr, Ast, mscore);
    // 6) gating
    gate_kernel<<<(M + 255) / 256, 256, 0, stream>>>(mscore, glin, mem_scale, gated);
    // 7) fuse seq_out + gated*mem_val -> bf16
    combine_kernel<<<(int)(((size_t)M * D_ + 255) / 256), 256, 0, stream>>>(seq, memval,
                                                                            gated, comb);
    // 8) output projection (WMMA) -> d_out
    gemm(comb, owb, out_b, out, M, D_, D_, D_);
}
